// KNNGenerator_54331336294752
// MI455X (gfx1250) — matrix operand order repaired and checked by reading, not compiled, not run on hardware
//
#include <hip/hip_runtime.h>
#include <hip/hip_bf16.h>
#include <stdint.h>

// ---------------------------------------------------------------------------
// KNN reconstruct for MI455X (gfx1250):
//   score(n,j) = ||a_j||^2 - 2 f_n . a_j   (rank-equivalent to L2 cdist)
//   kernel 0: f32 -> bf16 (feat pre-scaled by -2) + anchor norms  -> ws
//   kernel 1: WMMA bf16 GEMM (16x16x32) fused with per-row top-10.
//             Each wave owns 2 row-tiles x 1 col-tile (one shared B fragment
//             per 2 WMMAs -> 3 ds_load_b128 per WMMA instead of 4).
//             Anchor tiles streamed into LDS by the Tensor Data Mover
//             (double buffered, s_wait_tensorcnt), padded LDS pitch for
//             conflict-free b128 reads.
//   kernel 2: gather target_anchor rows by top-10 indices, mean -> out
// ---------------------------------------------------------------------------

typedef __bf16 v16bf __attribute__((ext_vector_type(16)));
typedef float  v8f   __attribute__((ext_vector_type(8)));
typedef unsigned int u32x4 __attribute__((ext_vector_type(4)));
typedef int          i32x8 __attribute__((ext_vector_type(8)));
typedef int          i32x4 __attribute__((ext_vector_type(4)));

#define N_FEAT 4096
#define M_ANCH 32768
#define DIM    512
#define KSEL   10

#define ROWS_PER_BLK   64            // feat rows per workgroup (4 row-tiles)
#define COLS_PER_CHUNK 64            // anchor cols per iteration (4 col-tiles)
#define N_CHUNKS       (M_ANCH / COLS_PER_CHUNK)
#define LDS_PITCH_E    520           // bf16 elems per LDS row (1040B, 260 dw; 260%64==4)
#define LDS_PITCH_B    (LDS_PITCH_E * 2)
#define SCORE_PITCH    68            // f32 elems per score row (68%64==4)

// round-to-nearest-even f32 -> bf16 (as raw u16)
__device__ __forceinline__ unsigned short f2bf_rne(float f) {
  unsigned u = __float_as_uint(f);
  unsigned r = (u + 0x7FFFu + ((u >> 16) & 1u)) >> 16;
  return (unsigned short)r;
}

// ---------------------------------------------------------------------------
// Kernel 0: convert anchors (+norms) and feats (scaled by -2) to bf16 in ws.
// blockIdx < M_ANCH: one anchor row; else: one feat row.
// ---------------------------------------------------------------------------
__global__ void knn_convert_kernel(const float* __restrict__ feat,
                                   const float* __restrict__ anchor,
                                   unsigned short* __restrict__ featB,
                                   unsigned short* __restrict__ anchorB,
                                   float* __restrict__ a2) {
  __shared__ float red[256];
  const int tid = threadIdx.x;
  const int row = blockIdx.x;
  if (row < M_ANCH) {
    float s = 0.f;
    for (int c = tid; c < DIM; c += 256) {
      float v = anchor[(size_t)row * DIM + c];
      anchorB[(size_t)row * DIM + c] = f2bf_rne(v);
      s += v * v;
    }
    red[tid] = s;
    __syncthreads();
    for (int st = 128; st > 0; st >>= 1) {
      if (tid < st) red[tid] += red[tid + st];
      __syncthreads();
    }
    if (tid == 0) a2[row] = red[0];
  } else {
    const int r = row - M_ANCH;
    for (int c = tid; c < DIM; c += 256)
      featB[(size_t)r * DIM + c] = f2bf_rne(-2.0f * feat[(size_t)r * DIM + c]);
  }
}

// ---------------------------------------------------------------------------
// TDM: load one 64x512 bf16 anchor tile (row-major, stride 512) into LDS,
// inserting 16B of padding after every 1024B row -> LDS pitch 1040B.
// D# group1 dword0: data_size=1(2B)<<16 | pad_enable<<20 | pad_interval=7<<22
//                   (256 dw = 1024B) | pad_amount=3<<25 (4 dw = 16B)
// ---------------------------------------------------------------------------
__device__ __forceinline__ void tdm_load_anchor_tile(unsigned ldsByteOff,
                                                     const void* gtile) {
  const uint64_t ga = (uint64_t)(uintptr_t)gtile;
  u32x4 g0 = { 1u,                                    // count=1 descriptor
               ldsByteOff,                            // lds_addr
               (unsigned)ga,                          // global_addr[31:0]
               (unsigned)(ga >> 32) | (2u << 30) };   // global_addr[56:32] | type=2
  i32x8 g1 = { (int)0x07D10000,                       // dsize/pad fields (see above)
               (int)(DIM << 16),                      // tensor_dim0 = 512 (low16 @ [31:16])
               (int)0x80000000,                       // tensor_dim1 = 32768 (low16 @ [31:16])
               (int)(DIM << 16),                      // tile_dim0 = 512 @ [31:16]
               COLS_PER_CHUNK,                        // tile_dim1 = 64 (tile_dim2 = 0 -> 2D)
               DIM,                                   // tensor_dim0_stride = 512
               0, 0 };
  i32x4 gz4 = { 0, 0, 0, 0 };
  i32x8 gz8 = { 0, 0, 0, 0, 0, 0, 0, 0 };
  // 6-arg form (clang-23 / therock-10.0 toolchain)
  __builtin_amdgcn_tensor_load_to_lds(g0, g1, gz4, gz4, gz8, 0);
}

// ---------------------------------------------------------------------------
// Kernel 1: fused bf16 WMMA GEMM + per-row top-10.
// 256 threads = 8 waves: wave w -> row-tile pair 2*(w>>2), col-tile (w&3)*16.
// ---------------------------------------------------------------------------
struct __align__(16) KnnSmem {
  unsigned short anchorBuf[2][COLS_PER_CHUNK * LDS_PITCH_E];  // 2 x 66560 B
  unsigned short featT[ROWS_PER_BLK * LDS_PITCH_E];           // 66560 B
  float scores[ROWS_PER_BLK * SCORE_PITCH];                   // 17408 B
  float tks[ROWS_PER_BLK][KSEL];
  int   tki[ROWS_PER_BLK][KSEL];
};

__global__ __launch_bounds__(256) void knn_topk_kernel(
    const unsigned short* __restrict__ featB,
    const unsigned short* __restrict__ anchorB,
    const float* __restrict__ a2,
    int* __restrict__ outIdx) {
  __shared__ KnnSmem sm;
  const int tid  = threadIdx.x;
  const int wid  = tid >> 5;
  const int lane = tid & 31;
  const int rtA  = (wid >> 2) * 2;    // row tiles rtA, rtA+1  (0..3)
  const int c0   = (wid & 3) << 4;    // col tile base 0/16/32/48
  const int half = lane >> 4;         // K-half select
  const int ln   = lane & 15;
  const int row0 = blockIdx.x * ROWS_PER_BLK;

  // LDS byte offsets of the two anchor buffers (flat addr[31:0] == LDS offset)
  const unsigned buf0Off = (unsigned)(uintptr_t)(void*)&sm.anchorBuf[0][0];
  const unsigned buf1Off = (unsigned)(uintptr_t)(void*)&sm.anchorBuf[1][0];

  // Kick off the first anchor tile DMA (wave 0 only; TDM ignores EXEC).
  if (wid == 0) tdm_load_anchor_tile(buf0Off, anchorB);

  // Stage the 64x512 bf16 feat slab into LDS (padded pitch), 16B chunks.
  {
    const uint4* fg = (const uint4*)(featB + (size_t)row0 * DIM);
    for (int i = tid; i < ROWS_PER_BLK * (DIM / 8); i += 256) {
      const int r = i >> 6;        // 64 x 16B chunks per row
      const int c = i & 63;
      *(uint4*)((char*)sm.featT + r * LDS_PITCH_B + c * 16) = fg[r * 64 + c];
    }
  }
  if (tid < ROWS_PER_BLK) {
    for (int k = 0; k < KSEL; ++k) { sm.tks[tid][k] = 3.0e38f; sm.tki[tid][k] = 0; }
  }

  for (int ch = 0; ch < N_CHUNKS; ++ch) {
    const int cur = ch & 1;
    if (wid == 0) {
      if (ch + 1 < N_CHUNKS) {
        tdm_load_anchor_tile(cur ? buf0Off : buf1Off,
                             anchorB + (size_t)(ch + 1) * COLS_PER_CHUNK * DIM);
        __builtin_amdgcn_s_wait_tensorcnt(1);   // tile `ch` complete (in-order)
      } else {
        __builtin_amdgcn_s_wait_tensorcnt(0);
      }
    }
    __syncthreads();

    // ---- two 16x16 score tiles: C init = a2[col]; A = (-2f) bf16; B = anchor
    v8f acc0, acc1;
    {
      const float a2v = a2[ch * COLS_PER_CHUNK + c0 + ln];
      #pragma unroll
      for (int v = 0; v < 8; ++v) { acc0[v] = a2v; acc1[v] = a2v; }
    }
    const uint4* fb0 = (const uint4*)((const char*)sm.featT +
                                      (rtA * 16 + ln) * LDS_PITCH_B);
    const uint4* fb1 = (const uint4*)((const char*)sm.featT +
                                      ((rtA + 1) * 16 + ln) * LDS_PITCH_B);
    const uint4* ab  = (const uint4*)((const char*)sm.anchorBuf[cur] +
                                      (c0 + ln) * LDS_PITCH_B);
    #pragma unroll
    for (int kk = 0; kk < 16; ++kk) {
      union { v16bf v; uint4 q[2]; } A0, A1, B;
      // A frags: lane row = rt*16+ln; K = kk*32 + half*8 + [0..8) and +16
      A0.q[0] = fb0[kk * 4 + half];
      A0.q[1] = fb0[kk * 4 + half + 2];
      A1.q[0] = fb1[kk * 4 + half];
      A1.q[1] = fb1[kk * 4 + half + 2];
      // B frag (shared by both WMMAs): lane col = c0+ln; K = kk*32 + half*8 + [0..8) and +16
      B.q[0] = ab[kk * 4 + half];
      B.q[1] = ab[kk * 4 + half + 2];
      acc0 = __builtin_amdgcn_wmma_f32_16x16x32_bf16(
          false, A0.v, false, B.v, (short)0, acc0, false, false);
      acc1 = __builtin_amdgcn_wmma_f32_16x16x32_bf16(
          false, A1.v, false, B.v, (short)0, acc1, false, false);
    }
    // C/D layout: VGPR v, lane L -> row v + 8*(L/16), col L%16
    #pragma unroll
    for (int v = 0; v < 8; ++v) {
      sm.scores[(rtA * 16 + half * 8 + v) * SCORE_PITCH + c0 + ln] = acc0[v];
      sm.scores[((rtA + 1) * 16 + half * 8 + v) * SCORE_PITCH + c0 + ln] = acc1[v];
    }
    __syncthreads();

    // ---- per-row top-10 maintenance (thread r owns row r)
    if (tid < ROWS_PER_BLK) {
      float worst = sm.tks[tid][KSEL - 1];
      const int base = ch * COLS_PER_CHUNK;
      for (int cc = 0; cc < COLS_PER_CHUNK; ++cc) {
        const float s = sm.scores[tid * SCORE_PITCH + cc];
        if (s < worst) {
          int j = KSEL - 1;
          while (j > 0 && sm.tks[tid][j - 1] > s) {
            sm.tks[tid][j] = sm.tks[tid][j - 1];
            sm.tki[tid][j] = sm.tki[tid][j - 1];
            --j;
          }
          sm.tks[tid][j] = s;
          sm.tki[tid][j] = base + cc;
          worst = sm.tks[tid][KSEL - 1];
        }
      }
    }
    __syncthreads();
  }

  if (tid < ROWS_PER_BLK) {
    for (int k = 0; k < KSEL; ++k)
      outIdx[(size_t)(row0 + tid) * KSEL + k] = sm.tki[tid][k];
  }
}

// ---------------------------------------------------------------------------
// Kernel 2: out[n] = mean over top-10 of target_anchor[idx]
// ---------------------------------------------------------------------------
__global__ void knn_gather_mean_kernel(const float* __restrict__ target,
                                       const int* __restrict__ idxK,
                                       float* __restrict__ out) {
  const int n = blockIdx.x;
  const int tid = threadIdx.x;
  int idx[KSEL];
  #pragma unroll
  for (int k = 0; k < KSEL; ++k) idx[k] = idxK[(size_t)n * KSEL + k];
  for (int c = tid; c < DIM; c += 256) {
    float s = 0.f;
    #pragma unroll
    for (int k = 0; k < KSEL; ++k) s += target[(size_t)idx[k] * DIM + c];
    out[(size_t)n * DIM + c] = s * (1.0f / KSEL);
  }
}

// ---------------------------------------------------------------------------
extern "C" void kernel_launch(void* const* d_in, const int* in_sizes, int n_in,
                              void* d_out, int out_size, void* d_ws, size_t ws_size,
                              hipStream_t stream) {
  (void)in_sizes; (void)n_in; (void)out_size; (void)ws_size;
  const float* feat   = (const float*)d_in[0];
  const float* anchor = (const float*)d_in[1];
  const float* target = (const float*)d_in[2];

  char* ws = (char*)d_ws;
  const size_t anchorB_off = 0;
  const size_t featB_off   = anchorB_off + (size_t)M_ANCH * DIM * 2;   // 32 MB
  const size_t a2_off      = featB_off   + (size_t)N_FEAT * DIM * 2;   // +4 MB
  const size_t idx_off     = a2_off      + (size_t)M_ANCH * 4;         // +128 KB
  unsigned short* anchorB = (unsigned short*)(ws + anchorB_off);
  unsigned short* featB   = (unsigned short*)(ws + featB_off);
  float*          a2      = (float*)(ws + a2_off);
  int*            idxK    = (int*)(ws + idx_off);

  knn_convert_kernel<<<M_ANCH + N_FEAT, 256, 0, stream>>>(feat, anchor, featB,
                                                          anchorB, a2);
  knn_topk_kernel<<<N_FEAT / ROWS_PER_BLK, 256, 0, stream>>>(featB, anchorB,
                                                             a2, idxK);
  knn_gather_mean_kernel<<<N_FEAT, 256, 0, stream>>>(target, idxK,
                                                     (float*)d_out);
}
